// GroupedQueryAttentionWithoutRoPE_60378650247248
// MI455X (gfx1250) — matrix operand order repaired and checked by reading, not compiled, not run on hardware
//
#include <hip/hip_runtime.h>
#include <hip/hip_bf16.h>
#include <stdint.h>

typedef _Float16 v16h __attribute__((ext_vector_type(16)));
typedef _Float16 v8h  __attribute__((ext_vector_type(8)));
typedef float    v8f  __attribute__((ext_vector_type(8)));
typedef unsigned int v4u __attribute__((ext_vector_type(4)));
typedef int      v8i __attribute__((ext_vector_type(8)));
typedef int      v4i __attribute__((ext_vector_type(4)));

#define WMMA_F32_F16(a, b, c) \
    __builtin_amdgcn_wmma_f32_16x16x32_f16(false, (a), false, (b), (short)0, (c), false, false)

static constexpr int BATCH = 2, SEQ = 1024, DMODEL = 2048;
static constexpr int NGROUPS = 8, HGRP = 4, HD = 64, DKV = 512;
static constexpr int MTOT = BATCH * SEQ;  // 2048 rows

// gfx1250 async copy: global memory -> LDS, 16B per active lane, ASYNCcnt-tracked.
__device__ __forceinline__ void async_b128(uint32_t lds_off, uint64_t gaddr) {
  asm volatile("global_load_async_to_lds_b128 %0, %1, off"
               :: "v"(lds_off), "v"(gaddr) : "memory");
}
__device__ __forceinline__ void wait_async0() {
  asm volatile("s_wait_asynccnt 0x0" ::: "memory");
}

// gfx1250 Tensor Data Mover: DMA a 2D tile (f16 elements) from a strided
// global tensor into contiguous LDS. D# per CDNA5 ISA.
// Issued once per executing wave (EXEC ignored); tracked by TENSORcnt.
__device__ __forceinline__ void tdm_load_2d_f16(
    uint32_t lds_addr, const void* gptr,
    uint32_t tensor_d0, uint32_t tensor_d1,   // tensor extents (elements)
    uint32_t tile_d0, uint32_t tile_d1,       // tile extents (elements)
    uint32_t stride0)                          // row stride (elements)
{
  const uint64_t ga = (uint64_t)(uintptr_t)gptr;
  v4u g0;
  g0.x = 1u;                                  // count=1 (valid user descriptor)
  g0.y = lds_addr;                            // lds_addr [63:32]
  g0.z = (uint32_t)ga;                        // global_addr[31:0]  -> bits 95:64
  g0.w = (uint32_t)((ga >> 32) & 0x1FFFFFFu)  // global_addr[56:32] -> bits 120:96
         | (2u << 30);                        // type=2 ("image")   -> bits 127:126
  v8i g1;
  g1[0] = (int)(1u << 16);                    // wg_mask=0, data_size=1 (2 bytes)
  g1[1] = (int)((tensor_d0 & 0xFFFFu) << 16); // tensor_dim0 low16  -> bits 63:48
  g1[2] = (int)((tensor_d0 >> 16) | ((tensor_d1 & 0xFFFFu) << 16)); // d0 hi / d1 lo
  g1[3] = (int)((tensor_d1 >> 16) | (tile_d0 << 16));               // d1 hi / tile0
  g1[4] = (int)tile_d1;                       // tile_dim1; tile_dim2=0
  g1[5] = (int)stride0;                       // tensor_dim0_stride low 32
  g1[6] = 0;                                  // stride0 hi / stride1 lo
  g1[7] = 0;                                  // stride1 hi
  v4i g2 = {0, 0, 0, 0};
  v4i g3 = {0, 0, 0, 0};
#if defined(__clang_major__) && (__clang_major__ >= 23)
  v8i g4 = {0, 0, 0, 0, 0, 0, 0, 0};
  __builtin_amdgcn_tensor_load_to_lds(g0, g1, g2, g3, g4, 0);
#else
  __builtin_amdgcn_tensor_load_to_lds(g0, g1, g2, g3, 0);
#endif
}
__device__ __forceinline__ void wait_tensor0() {
  __builtin_amdgcn_s_wait_tensorcnt((short)0);
}

// ---------------------------------------------------------------------------
// Tiled GEMM: C[M,N] = A[M,K] @ W[N,K]^T   (torch Linear convention)
// BM=128, BN=64, BK=32; 256 threads = 8 waves; each wave computes a 16x64 strip.
// AT = float (q/k/v proj, staged+converted through VGPRs) or _Float16
// (o proj, staged via async-to-LDS byte copies). OUT_F32 adds bias, writes fp32.
// ---------------------------------------------------------------------------
template <typename AT, bool OUT_F32>
__global__ __launch_bounds__(256) void gemm_rt_kernel(
    const AT* __restrict__ A, const float* __restrict__ W, void* __restrict__ Cv,
    const float* __restrict__ bias, int M, int N, int K) {
  constexpr int BM = 128, BN = 64, BK = 32;
  __shared__ _Float16 As[BM][BK];
  __shared__ _Float16 Bs[BN][BK];

  const int tid = threadIdx.x;
  const int wave = tid >> 5, lane = tid & 31;
  const int nl = lane & 15, hi = lane >> 4;
  const int m0 = blockIdx.y * BM, n0 = blockIdx.x * BN;

  v8f zero = {};
  v8f acc[4];
#pragma unroll
  for (int j = 0; j < 4; ++j) acc[j] = zero;

  for (int k0 = 0; k0 < K; k0 += BK) {
    // ---- stage A tile ----
    if constexpr (sizeof(AT) == 4) {
#pragma unroll
      for (int it = 0; it < (BM * BK) / (256 * 4); ++it) {
        int e = (it * 256 + tid) * 4;
        int r = e / BK, c = e % BK;
        const float4 f = *(const float4*)(A + (size_t)(m0 + r) * K + k0 + c);
        As[r][c + 0] = (_Float16)f.x;
        As[r][c + 1] = (_Float16)f.y;
        As[r][c + 2] = (_Float16)f.z;
        As[r][c + 3] = (_Float16)f.w;
      }
    } else {
      // f16 A: pure byte copy -> async global->LDS, no VGPR round-trip
#pragma unroll
      for (int it = 0; it < (BM * BK) / (256 * 8); ++it) {
        int e = (it * 256 + tid) * 8;
        int r = e / BK, c = e % BK;
        async_b128((uint32_t)(uintptr_t)&As[r][c],
                   (uint64_t)(uintptr_t)(A + (size_t)(m0 + r) * K + k0 + c));
      }
    }
    // ---- stage W tile: Bs[n][k] = W[n0+n][k0+k] ----
#pragma unroll
    for (int it = 0; it < (BN * BK) / (256 * 4); ++it) {
      int e = (it * 256 + tid) * 4;
      int r = e / BK, c = e % BK;
      const float4 f = *(const float4*)(W + (size_t)(n0 + r) * K + k0 + c);
      Bs[r][c + 0] = (_Float16)f.x;
      Bs[r][c + 1] = (_Float16)f.y;
      Bs[r][c + 2] = (_Float16)f.z;
      Bs[r][c + 3] = (_Float16)f.w;
    }
    // ---- prefetch next tiles into cache while this one is consumed ----
    if (k0 + BK < K) {
      int e = tid * 4;
      int r = e / BK, c = e % BK;
      __builtin_prefetch(A + (size_t)(m0 + r) * K + (k0 + BK) + c, 0, 0);
      if (r < BN) __builtin_prefetch(W + (size_t)(n0 + r) * K + (k0 + BK) + c, 0, 0);
    }
    if constexpr (sizeof(AT) == 2) wait_async0();
    __syncthreads();

    // ---- A fragment: 16x32, lane=M, K halves split by lane-half ----
    v16h a;
#pragma unroll
    for (int h = 0; h < 16; ++h) {
      int kk = hi * 16 + h;
      a[h] = As[wave * 16 + nl][kk];
    }
#pragma unroll
    for (int j = 0; j < 4; ++j) {
      // B fragment: 32x16, lane=N (=col of W^T), K-half selected by lane-half
      v16h b;
#pragma unroll
      for (int h = 0; h < 16; ++h) b[h] = Bs[j * 16 + nl][hi * 16 + h];
      acc[j] = WMMA_F32_F16(a, b, acc[j]);
    }
    __syncthreads();
  }

  // ---- store: C layout lane=N, rows r + 8*hi ----
#pragma unroll
  for (int j = 0; j < 4; ++j) {
#pragma unroll
    for (int r = 0; r < 8; ++r) {
      int row = m0 + wave * 16 + r + hi * 8;
      int col = n0 + j * 16 + nl;
      if constexpr (OUT_F32)
        ((float*)Cv)[(size_t)row * N + col] = acc[j][r] + bias[col];
      else
        ((_Float16*)Cv)[(size_t)row * N + col] = (_Float16)acc[j][r];
    }
  }
}

// ---------------------------------------------------------------------------
// Flash attention per head (causal + padding mask), GQA: head (g,h) uses K/V
// of group g. 128 threads = 4 waves; each wave owns 16 query rows; workgroup
// covers 64 rows of one head. Double-buffered 32x64 K/V tiles DMA'd into LDS
// by the Tensor Data Mover (wave 0 issues, TENSORcnt-tracked); online softmax;
// WMMA for QK^T and PV.
// ---------------------------------------------------------------------------
__global__ __launch_bounds__(128) void gqa_attn_kernel(
    const _Float16* __restrict__ qf, const _Float16* __restrict__ kf,
    const _Float16* __restrict__ vf, const int* __restrict__ amask,
    _Float16* __restrict__ outp) {
  __shared__ _Float16 Ks[2][32][64];
  __shared__ _Float16 Vs[2][32][64];
  __shared__ _Float16 Ps[4][16][32];

  const int tid = threadIdx.x;
  const int wave = tid >> 5, lane = tid & 31;
  const int nl = lane & 15, hi = lane >> 4;
  const int head = blockIdx.y;
  const int b = head >> 5, rem = head & 31, g = rem >> 2, hq = rem & 3;
  const int qbase = blockIdx.x * 64 + wave * 16;
  const int doff = (g * HGRP + hq) * HD;  // offset in D=2048 q/out rows
  const int dkv = g * HD;                 // offset in 512-wide k/v rows

  // ---- Q fragments for this wave's 16 rows (two 32-k chunks of HD=64) ----
  v16h qa[2];
  {
    const _Float16* qrow = qf + (size_t)(b * SEQ + qbase + nl) * DMODEL + doff;
#pragma unroll
    for (int c = 0; c < 2; ++c)
#pragma unroll
      for (int h = 0; h < 16; ++h) {
        int kk = c * 32 + hi * 16 + h;
        qa[c][h] = qrow[kk];
      }
  }

  v8f zero = {};
  v8f oacc[4];
#pragma unroll
  for (int nt = 0; nt < 4; ++nt) oacc[nt] = zero;
  float mrun[8], lrun[8];
#pragma unroll
  for (int r = 0; r < 8; ++r) { mrun[r] = -3.0e38f; lrun[r] = 0.f; }

  // TDM issue of one 32-key K/V tile pair into LDS buffer bb (wave 0 only:
  // TDM ignores EXEC, so every issuing wave would start its own DMA).
  auto issue_tile = [&](int j, int bb) {
    if (wave == 0) {
      const int kb = j * 32;
      const size_t elem = (size_t)(b * SEQ + kb) * DKV + dkv;
      tdm_load_2d_f16((uint32_t)(uintptr_t)&Ks[bb][0][0], kf + elem,
                      DKV, (uint32_t)MTOT, HD, 32, DKV);
      tdm_load_2d_f16((uint32_t)(uintptr_t)&Vs[bb][0][0], vf + elem,
                      DKV, (uint32_t)MTOT, HD, 32, DKV);
    }
  };

  const int nb = blockIdx.x * 2 + 2;  // key blocks needed by this workgroup
  issue_tile(0, 0);

  for (int j = 0; j < nb; ++j) {
    const int kb = j * 32;
    const int bb = j & 1;
    if (wave == 0) wait_tensor0();   // tile j's DMA landed in LDS
    __syncthreads();                 // publish tile j to all waves
    if (j + 1 < nb) issue_tile(j + 1, (j + 1) & 1);  // overlap next tile

    if (kb <= qbase + 15) {  // tile not fully causally masked for this wave
      // ---- S = Q K^T / sqrt(64): two 16x16 score tiles over 32 keys ----
      v8f s[2];
#pragma unroll
      for (int st = 0; st < 2; ++st) {
        v8f sa = zero;
#pragma unroll
        for (int c = 0; c < 2; ++c) {
          v16h bk;  // B = K^T chunk: lane = key, element h -> head-dim
#pragma unroll
          for (int h = 0; h < 16; ++h)
            bk[h] = Ks[bb][st * 16 + nl][c * 32 + hi * 16 + h];
          sa = WMMA_F32_F16(qa[c], bk, sa);
        }
        const int keycol = kb + st * 16 + nl;
        const int mk = amask[b * SEQ + keycol];
#pragma unroll
        for (int r = 0; r < 8; ++r) {
          const int row = qbase + r + hi * 8;
          float v = sa[r] * 0.125f;
          if (keycol > row || mk == 0) v = -3.0e38f;
          sa[r] = v;
        }
        s[st] = sa;
      }
      // ---- online softmax: row reductions across the 16 lanes per half ----
      float tmax[8], alpha[8], tsum[8];
#pragma unroll
      for (int r = 0; r < 8; ++r) tmax[r] = fmaxf(s[0][r], s[1][r]);
#pragma unroll
      for (int d = 1; d < 16; d <<= 1)
#pragma unroll
        for (int r = 0; r < 8; ++r)
          tmax[r] = fmaxf(tmax[r], __shfl_xor(tmax[r], d, 32));
#pragma unroll
      for (int r = 0; r < 8; ++r) {
        float mn = fmaxf(mrun[r], tmax[r]);
        alpha[r] = __expf(mrun[r] - mn);
        mrun[r] = mn;
        tsum[r] = 0.f;
      }
#pragma unroll
      for (int st = 0; st < 2; ++st)
#pragma unroll
        for (int r = 0; r < 8; ++r) {
          float p = __expf(s[st][r] - mrun[r]);
          s[st][r] = p;
          tsum[r] += p;
        }
#pragma unroll
      for (int d = 1; d < 16; d <<= 1)
#pragma unroll
        for (int r = 0; r < 8; ++r) tsum[r] += __shfl_xor(tsum[r], d, 32);
#pragma unroll
      for (int r = 0; r < 8; ++r) lrun[r] = lrun[r] * alpha[r] + tsum[r];
#pragma unroll
      for (int nt = 0; nt < 4; ++nt)
#pragma unroll
        for (int r = 0; r < 8; ++r) oacc[nt][r] *= alpha[r];

      // ---- re-stripe P (C layout) into A-fragment layout via per-wave LDS ----
#pragma unroll
      for (int st = 0; st < 2; ++st)
#pragma unroll
        for (int r = 0; r < 8; ++r)
          Ps[wave][r + hi * 8][st * 16 + nl] = (_Float16)s[st][r];
      asm volatile("s_wait_dscnt 0x0" ::: "memory");
      v16h pa;
#pragma unroll
      for (int h = 0; h < 16; ++h) {
        int kk = hi * 16 + h;
        pa[h] = Ps[wave][nl][kk];
      }
      // ---- O += P @ V: four 16-wide d-tiles ----
#pragma unroll
      for (int nt = 0; nt < 4; ++nt) {
        v16h vb;  // B = V chunk: lane = d, element h -> key
#pragma unroll
        for (int h = 0; h < 16; ++h) vb[h] = Vs[bb][hi * 16 + h][nt * 16 + nl];
        oacc[nt] = WMMA_F32_F16(pa, vb, oacc[nt]);
      }
    }
    __syncthreads();  // all reads of buffer bb done before it is refilled
  }

  // ---- normalize and write f16 head output into (B,S,D) layout ----
  float invl[8];
#pragma unroll
  for (int r = 0; r < 8; ++r) invl[r] = lrun[r] > 0.f ? 1.f / lrun[r] : 0.f;
#pragma unroll
  for (int nt = 0; nt < 4; ++nt)
#pragma unroll
    for (int r = 0; r < 8; ++r) {
      int row = qbase + r + hi * 8;
      outp[(size_t)(b * SEQ + row) * DMODEL + doff + nt * 16 + nl] =
          (_Float16)(oacc[nt][r] * invl[r]);
    }
}

// ---------------------------------------------------------------------------
extern "C" void kernel_launch(void* const* d_in, const int* in_sizes, int n_in,
                              void* d_out, int out_size, void* d_ws, size_t ws_size,
                              hipStream_t stream) {
  const float* Q  = (const float*)d_in[0];
  const float* K  = (const float*)d_in[1];
  const float* V  = (const float*)d_in[2];
  const int*   am = (const int*)d_in[3];
  const float* wq = (const float*)d_in[4];
  const float* wk = (const float*)d_in[5];
  const float* wv = (const float*)d_in[6];
  const float* wo = (const float*)d_in[7];
  const float* bo = (const float*)d_in[8];

  char* ws = (char*)d_ws;
  const size_t qbytes = (size_t)MTOT * DMODEL * sizeof(_Float16);  // 8 MiB
  const size_t kbytes = (size_t)MTOT * DKV * sizeof(_Float16);     // 2 MiB
  _Float16* qf   = (_Float16*)(ws);
  _Float16* kfp  = (_Float16*)(ws + qbytes);
  _Float16* vfp  = (_Float16*)(ws + qbytes + kbytes);
  _Float16* attn = (_Float16*)(ws + qbytes + 2 * kbytes);

  // q/k/v projections (fp32 in, f16 out)
  gemm_rt_kernel<float, false><<<dim3(DMODEL / 64, MTOT / 128), 256, 0, stream>>>(
      Q, wq, qf, nullptr, MTOT, DMODEL, DMODEL);
  gemm_rt_kernel<float, false><<<dim3(DKV / 64, MTOT / 128), 256, 0, stream>>>(
      K, wk, kfp, nullptr, MTOT, DKV, DMODEL);
  gemm_rt_kernel<float, false><<<dim3(DKV / 64, MTOT / 128), 256, 0, stream>>>(
      V, wv, vfp, nullptr, MTOT, DKV, DMODEL);

  // flash attention: grid = (query blocks of 64, B*G*HG heads)
  gqa_attn_kernel<<<dim3(SEQ / 64, BATCH * NGROUPS * HGRP), 128, 0, stream>>>(
      qf, kfp, vfp, am, attn);

  // output projection with bias (f16 in, fp32 out)
  gemm_rt_kernel<_Float16, true><<<dim3(DMODEL / 64, MTOT / 128), 256, 0, stream>>>(
      attn, wo, (float*)d_out, bo, MTOT, DMODEL, DMODEL);
}
